// AlignHead_824633721410
// MI455X (gfx1250) — matrix operand order repaired and checked by reading, not compiled, not run on hardware
//
#include <hip/hip_runtime.h>
#include <hip/hip_bf16.h>

typedef __attribute__((ext_vector_type(16))) __bf16 v16bf;
typedef __attribute__((ext_vector_type(8)))  __bf16 v8bf;
typedef __attribute__((ext_vector_type(2)))  __bf16 v2bf;
typedef __attribute__((ext_vector_type(8)))  float  v8f;

#define CC    256
#define HH    64
#define WW    64
#define PLANE (HH * WW)
#define KTOT  2304    // 9 * 256
#define NCHUNK 72     // KTOT / 32

// ---------------------------------------------------------------------------
// Weight prep: w_def (O,C,3,3) f32 -> bf16 in WMMA B-fragment layout.
//   k = tap*256 + c ; chunk = k/32, kk = k%32
//   lane = (kk/16)*16 | (o%16), j = kk%16
//   wfrag[((chunk*16 + o/16)*32 + lane)*16 + j]
// ---------------------------------------------------------------------------
__global__ void prep_weights_kernel(const float* __restrict__ wdef,
                                    __bf16* __restrict__ wfrag) {
    int idx = blockIdx.x * 256 + threadIdx.x;   // 2304*256 threads
    int k = idx >> 8;
    int o = idx & 255;
    int tap = k >> 8;
    int c   = k & 255;
    float v = wdef[(size_t)o * 2304 + c * 9 + tap];
    int chunk = k >> 5;
    int kk    = k & 31;
    int lane  = ((kk >> 4) << 4) | (o & 15);
    int j     = kk & 15;
    int otile = o >> 4;
    wfrag[((((size_t)chunk * 16 + otile) * 32 + lane) << 4) + j] = (__bf16)v;
}

// ---------------------------------------------------------------------------
// Fused deformable-conv3x3 + ReLU + 1x1 head.
// One WG = 32 pixels (two M-tiles); 8 waves x 2 o-tiles cover O=256.
// 4 LDS A-buffers, 2-chunk pipeline stages: 8 WMMAs per barrier,
// all gathers / B-loads are immediate-offset VMEM ops.
// ---------------------------------------------------------------------------
__global__ void __launch_bounds__(256)
deform_head_kernel(const float* __restrict__ x,
                   const float* __restrict__ off,
                   const float* __restrict__ wcls,
                   const float* __restrict__ bcls,
                   const __bf16* __restrict__ wfrag,
                   float* __restrict__ out)
{
    __shared__ __align__(16) __bf16 Abuf[4][2][16 * 32]; // [buf][mtile][px*32+k]
    __shared__ __align__(16) int    sIdx[9][32][4];
    __shared__ __align__(16) float  sWgt[9][32][4];
    __shared__ float part[32];

    const int tid  = threadIdx.x;
    const int lane = tid & 31;
    const int wave = tid >> 5;

    const int pbase = blockIdx.x << 5;        // 32 pixels, same image row
    const int n     = pbase >> 12;
    const int h     = (pbase >> 6) & 63;
    const int wbase = pbase & 63;             // multiple of 32

    const float* ximg = x + (size_t)n * CC * PLANE;

    // ---- bilinear corner metadata for all 9 taps, once ----
    for (int idx = tid; idx < 9 * 32; idx += 256) {
        int tap = idx >> 5;
        int px  = idx & 31;
        int ky = tap / 3, kx = tap - 3 * ky;
        const float* op = off + ((size_t)(pbase + px)) * 18 + 2 * tap;
        float py  = (float)(h + ky - 1) + op[0];
        float pxx = (float)(wbase + px + kx - 1) + op[1];
        float y0f = floorf(py), x0f = floorf(pxx);
        float wy1 = py - y0f,  wx1 = pxx - x0f;
        int y0 = (int)y0f, x0 = (int)x0f;
#pragma unroll
        for (int a = 0; a < 2; ++a)
#pragma unroll
            for (int b = 0; b < 2; ++b) {
                int yy = y0 + a, xx = x0 + b;
                bool valid = (yy >= 0) & (yy < HH) & (xx >= 0) & (xx < WW);
                int yc = min(max(yy, 0), HH - 1);
                int xc = min(max(xx, 0), WW - 1);
                sIdx[tap][px][a * 2 + b] = yc * WW + xc;
                float wa = a ? wy1 : 1.0f - wy1;
                float wb = b ? wx1 : 1.0f - wx1;
                sWgt[tap][px][a * 2 + b] = valid ? wa * wb : 0.0f;
            }
    }
    __syncthreads();

    // sampling assignment: 32 consecutive threads walk 32 consecutive pixels
    const int spx = tid & 31;       // pixel within WG tile
    const int scp = tid >> 5;       // c-pair group 0..7
    const int smt = spx >> 4;       // mtile of this pixel
    const int spl = spx & 15;       // pixel within mtile

    // per-tap gather state: 4 corner base pointers (fold in 2*scp channels)
    const float* p0; const float* p1; const float* p2; const float* p3;
    float4 wv;

    auto setup_tap = [&](int tap) {
        int4 iv = *(const int4*)&sIdx[tap][spx][0];
        wv      = *(const float4*)&sWgt[tap][spx][0];
        const float* base = ximg + (size_t)(2 * scp) * PLANE;
        p0 = base + iv.x; p1 = base + iv.y;
        p2 = base + iv.z; p3 = base + iv.w;
    };

    // sub compile-time after unrolling -> all gathers are imm-offset loads
    auto sample_sub = [&](int sub, int buf) {
        __bf16* arow = &Abuf[buf][smt][spl * 32];
#pragma unroll
        for (int pass = 0; pass < 2; ++pass) {
            const int e = (sub * 32 + pass * 16) * PLANE;  // element offset
            float g0 = p0[e];         float g1 = p1[e];
            float g2 = p2[e];         float g3 = p3[e];
            float g4 = p0[e + PLANE]; float g5 = p1[e + PLANE];
            float g6 = p2[e + PLANE]; float g7 = p3[e + PLANE];
            float v0 = g0 * wv.x + g1 * wv.y + g2 * wv.z + g3 * wv.w;
            float v1 = g4 * wv.x + g5 * wv.y + g6 * wv.z + g7 * wv.w;
            v2bf pk;
            pk[0] = (__bf16)v0;
            pk[1] = (__bf16)v1;
            *(v2bf*)&arow[2 * (scp + 8 * pass)] = pk;      // one ds_store_b32
        }
    };

    // WMMA fragment addressing
    const int m      = lane & 15;
    const int khalf  = lane >> 4;
    const int otile0 = wave;
    const int otile1 = wave + 8;

    v8f acc00 = {}, acc01 = {}, acc10 = {}, acc11 = {};

    auto load_afrag = [&](int buf, int mt) -> v16bf {
        const __bf16* ap = &Abuf[buf][mt][m * 32 + khalf * 16];
        v8bf lo = *(const v8bf*)ap;
        v8bf hi = *(const v8bf*)(ap + 8);
        return __builtin_shufflevector(lo, hi,
            0,1,2,3,4,5,6,7,8,9,10,11,12,13,14,15);
    };

    setup_tap(0);
    sample_sub(0, 0);
    sample_sub(1, 1);
    __syncthreads();

    // B pointer for this wave's otile0 block; advance once per tap.
    // byte offsets: chunk-in-tap -> sub*16384 ; otile1 -> +8192
    const char* bptr = (const char*)(wfrag +
        (((size_t)otile0 * 32 + lane) << 4));

    for (int tap = 0; tap < 9; ++tap) {
#pragma unroll
        for (int p = 0; p < 4; ++p) {
            const int s0 = 2 * p;          // two chunks per stage
            const int s1 = 2 * p + 1;
            const int buf0 = s0 & 3;       // (tap*8+s)&3 == s&3
            const int buf1 = s1 & 3;

            // fragments for both chunks of this stage (independent loads)
            v16bf a00 = load_afrag(buf0, 0);
            v16bf a01 = load_afrag(buf0, 1);
            v16bf a10 = load_afrag(buf1, 0);
            v16bf a11 = load_afrag(buf1, 1);
            v16bf b00 = *(const v16bf*)(bptr + s0 * 16384);
            v16bf b01 = *(const v16bf*)(bptr + s0 * 16384 + 8192);
            v16bf b10 = *(const v16bf*)(bptr + s1 * 16384);
            v16bf b11 = *(const v16bf*)(bptr + s1 * 16384 + 8192);

            // overlap: sample the next pair into the other two buffers
            if (p < 3) {
                sample_sub(s0 + 2, (s0 + 2) & 3);
                sample_sub(s1 + 2, (s1 + 2) & 3);
            } else if (tap < 8) {
                setup_tap(tap + 1);        // old gather pointers are dead here
                sample_sub(0, 0);
                sample_sub(1, 1);
            }

            acc00 = __builtin_amdgcn_wmma_f32_16x16x32_bf16(
                false, a00, false, b00, (short)0, acc00, false, false);
            acc10 = __builtin_amdgcn_wmma_f32_16x16x32_bf16(
                false, a01, false, b00, (short)0, acc10, false, false);
            acc01 = __builtin_amdgcn_wmma_f32_16x16x32_bf16(
                false, a00, false, b01, (short)0, acc01, false, false);
            acc11 = __builtin_amdgcn_wmma_f32_16x16x32_bf16(
                false, a01, false, b01, (short)0, acc11, false, false);
            acc00 = __builtin_amdgcn_wmma_f32_16x16x32_bf16(
                false, a10, false, b10, (short)0, acc00, false, false);
            acc10 = __builtin_amdgcn_wmma_f32_16x16x32_bf16(
                false, a11, false, b10, (short)0, acc10, false, false);
            acc01 = __builtin_amdgcn_wmma_f32_16x16x32_bf16(
                false, a10, false, b11, (short)0, acc01, false, false);
            acc11 = __builtin_amdgcn_wmma_f32_16x16x32_bf16(
                false, a11, false, b11, (short)0, acc11, false, false);

            __syncthreads();   // one barrier per 2 chunks (4 buffers)
        }
        bptr += 8 * 16384;     // next tap's chunk block
    }

    // ---- fused epilogue: relu, dot with w_cls over O, + bias ----
    if (tid < 32) part[tid] = bcls[0];
    __syncthreads();

    const int ncol = lane & 15;
    const float wc0 = wcls[otile0 * 16 + ncol];
    const float wc1 = wcls[otile1 * 16 + ncol];

    float s0[8], s1[8];
#pragma unroll
    for (int v = 0; v < 8; ++v) {
        float f;
        f = acc00[v] > 0.0f ? acc00[v] : 0.0f;  s0[v]  = f * wc0;
        f = acc01[v] > 0.0f ? acc01[v] : 0.0f;  s0[v] += f * wc1;
        f = acc10[v] > 0.0f ? acc10[v] : 0.0f;  s1[v]  = f * wc0;
        f = acc11[v] > 0.0f ? acc11[v] : 0.0f;  s1[v] += f * wc1;
    }
#pragma unroll
    for (int msk = 1; msk <= 8; msk <<= 1)
#pragma unroll
        for (int v = 0; v < 8; ++v) {
            s0[v] += __shfl_xor(s0[v], msk, 32);
            s1[v] += __shfl_xor(s1[v], msk, 32);
        }

    if (ncol == 0) {                       // lanes 0 and 16 hold pixel sums
        const int pb = khalf * 8;
#pragma unroll
        for (int v = 0; v < 8; ++v) {
            atomicAdd(&part[pb + v],      s0[v]);   // mtile0: pixels 0..15
            atomicAdd(&part[16 + pb + v], s1[v]);   // mtile1: pixels 16..31
        }
    }
    __syncthreads();
    if (tid < 32) out[pbase + tid] = part[tid];
}

extern "C" void kernel_launch(void* const* d_in, const int* in_sizes, int n_in,
                              void* d_out, int out_size, void* d_ws, size_t ws_size,
                              hipStream_t stream) {
    const float* x    = (const float*)d_in[0];   // (16,256,64,64)
    const float* off  = (const float*)d_in[1];   // (16,4096,18)
    const float* wdef = (const float*)d_in[2];   // (256,256,3,3)
    const float* wcls = (const float*)d_in[3];   // (1,256,1,1)
    const float* bcls = (const float*)d_in[4];   // (1,)
    float* out = (float*)d_out;                  // (16,1,64,64)

    __bf16* wfrag = (__bf16*)d_ws;               // 1.18 MB in workspace

    prep_weights_kernel<<<KTOT, 256, 0, stream>>>(wdef, wfrag);
    deform_head_kernel<<<2048, 256, 0, stream>>>(x, off, wcls, bcls, wfrag, out);
}
